// BERT_472446403109
// MI455X (gfx1250) — matrix operand order repaired and checked by reading, not compiled, not run on hardware
//
#include <hip/hip_runtime.h>
#include <cmath>

// ---------------------------------------------------------------------------
// BERT forward for MI455X (gfx1250, wave32, WMMA bf16 16x16x32, f32 accum).
//  - Weights pre-packed once to bf16, transposed [N][K]; GEMM weight tiles are
//    staged with global_load_async_to_lds_b64 (ASYNCcnt, no VGPR round trip).
//  - Activation tiles convert f32->bf16 in pairs (f32x2 load, packed u32 LDS
//    store) while staging.
//  - Attention scores are written straight into the d_out attention region and
//    softmaxed (scale+mask fused) in place.
// ---------------------------------------------------------------------------

static constexpr int Bc  = 8;
static constexpr int Sc  = 1024;
static constexpr int Dc  = 768;
static constexpr int Hc  = 12;
static constexpr int Lc  = 2;
static constexpr int FFc = 3072;
static constexpr int DKc = 64;
static constexpr int NT  = 256;   // threads per block (8 wave32 waves)

typedef __attribute__((ext_vector_type(16))) __bf16 v16bf;
typedef __attribute__((ext_vector_type(8)))  __bf16 v8bf;
typedef __attribute__((ext_vector_type(8)))  float  v8f;
typedef __attribute__((ext_vector_type(2)))  float  f32x2;

__device__ __forceinline__ __bf16 to_bf16(float f) {
  unsigned u = __builtin_bit_cast(unsigned, f);
  unsigned r = (u + 0x7FFFu + ((u >> 16) & 1u)) >> 16;   // round-to-nearest-even
  unsigned short s = (unsigned short)r;
  return __builtin_bit_cast(__bf16, s);
}

__device__ __forceinline__ unsigned pack2_bf16(float a, float b) {
  unsigned ua = __builtin_bit_cast(unsigned, a);
  unsigned ub = __builtin_bit_cast(unsigned, b);
  ua = (ua + 0x7FFFu + ((ua >> 16) & 1u)) >> 16;
  ub = (ub + 0x7FFFu + ((ub >> 16) & 1u)) & 0xFFFF0000u;
  return (ua & 0xFFFFu) | ub;
}

// ---------------------------------------------------------------------------
// One-time weight prepack: Wt[n*K + k] = bf16(W[k*N + n])
// ---------------------------------------------------------------------------
__global__ __launch_bounds__(256)
void prepack_wT_kernel(const float* __restrict__ W, __bf16* __restrict__ Wt,
                       int K, int N) {
  size_t i = (size_t)blockIdx.x * NT + threadIdx.x;
  int k = (int)(i / N), n = (int)(i % N);
  Wt[(size_t)n * K + k] = to_bf16(W[i]);
}

// ---------------------------------------------------------------------------
// Generic tiled GEMM: C[z] = act(A[z] @ B[z] + bias)
//   A: f32 [M x K].  B depends on BMODE:
//     BMODE 0: f32  [K x N]           (attn @ V; V is an f32 activation)
//     BMODE 1: f32  [N x K]           (Q @ K^T; both f32 activations)
//     BMODE 2: bf16 [N x K] prepacked (all weight GEMMs; async copy to LDS)
//   Batched over grid.z with zb = z / ZH, zh = z % ZH offsets (attention).
//   Wave tile 32x64 = 2x4 v_wmma_f32_16x16x32_bf16 accumulators; BK = 32.
// ---------------------------------------------------------------------------
template <int WMW, int WNW, int BMODE, int ACT>
__global__ __launch_bounds__(256)
void gemm_bf16_wmma(const float* __restrict__ A, int lda, long sAb, long sAh,
                    const void* __restrict__ Bptr, int ldb, long sBb, long sBh,
                    float* __restrict__ C, int ldc, long sCb, long sCh,
                    const float* __restrict__ bias, int K, int ZH) {
  constexpr int BM  = WMW * 32;
  constexpr int BN  = WNW * 64;
  constexpr int BK  = 32;
  constexpr int BKP = 48;     // padded LDS row (bf16 units) -> 96B, 16B aligned

  __shared__ __bf16 As[BM][BKP];
  __shared__ __bf16 Bs[BN][BKP];

  const int tid = threadIdx.x;
  const int z   = blockIdx.z;
  const int zb  = z / ZH;
  const int zh  = z - zb * ZH;

  const float* Ag = A + (size_t)zb * sAb + (size_t)zh * sAh
                      + (size_t)blockIdx.x * BM * lda;
  const float*  Bgf = (const float*)Bptr + (size_t)zb * sBb + (size_t)zh * sBh;
  const __bf16* Bgh = (const __bf16*)Bptr;
  float*        Cg  = C + (size_t)zb * sCb + (size_t)zh * sCh;
  const int n0 = blockIdx.y * BN;

  const int wave = tid >> 5;
  const int lane = tid & 31;
  const int wm   = wave / WNW;
  const int wn   = wave - wm * WNW;
  const int hl   = lane >> 4;     // which 16-lane half
  const int l16  = lane & 15;

  const unsigned bsbase = (unsigned)(uintptr_t)(&Bs[0][0]);

  v8f acc[2][4] = {};

  for (int k0 = 0; k0 < K; k0 += BK) {
    // ---- stage A tile (BM x 32): f32x2 loads, packed bf16x2 LDS stores ----
    for (int idx = tid; idx < BM * 16; idx += NT) {
      int m = idx >> 4, kp = idx & 15;
      f32x2 a2 = *(const f32x2*)&Ag[(size_t)m * lda + k0 + 2 * kp];
      *(unsigned*)&As[m][2 * kp] = pack2_bf16(a2.x, a2.y);
    }
    if (k0 + BK < K)
      __builtin_prefetch(&Ag[(size_t)(tid >> 1) * lda + k0 + BK], 0, 1);

    // ---- stage B tile -> LDS as Bs[n][k] ----
    if constexpr (BMODE == 2) {
      // prepacked bf16 [N][K]: pure async byte copy, 8B per lane-op
      for (int c = tid; c < BN * 8; c += NT) {
        int n = c >> 3, part = c & 7;
        const __bf16* gp = Bgh + (size_t)(n0 + n) * ldb + k0 + part * 4;
        unsigned laddr = bsbase + (unsigned)(n * (BKP * 2) + part * 8);
        asm volatile("global_load_async_to_lds_b64 %0, %1, off"
                     :: "v"(laddr), "v"((unsigned long long)(uintptr_t)gp)
                     : "memory");
      }
      asm volatile("s_wait_asynccnt 0x0" ::: "memory");
    } else if constexpr (BMODE == 1) {
      // f32 [N][K]: pairs contiguous along k on both sides
      for (int idx = tid; idx < BN * 16; idx += NT) {
        int n = idx >> 4, kp = idx & 15;
        f32x2 w = *(const f32x2*)&Bgf[(size_t)(n0 + n) * ldb + k0 + 2 * kp];
        *(unsigned*)&Bs[n][2 * kp] = pack2_bf16(w.x, w.y);
      }
    } else {
      // f32 [K][N]: pair along n (coalesced load), two b16 stores
      for (int idx = tid; idx < BK * (BN / 2); idx += NT) {
        int kk = idx / (BN / 2), np = idx - kk * (BN / 2);
        f32x2 w = *(const f32x2*)&Bgf[(size_t)(k0 + kk) * ldb + n0 + 2 * np];
        Bs[2 * np + 0][kk] = to_bf16(w.x);
        Bs[2 * np + 1][kk] = to_bf16(w.y);
      }
    }
    __syncthreads();

    // ---- load fragments (ISA 16-bit A 16x32 / B 32x16 lane layouts) ----
    v16bf afr[2];
#pragma unroll
    for (int mi = 0; mi < 2; ++mi) {
      const __bf16* ap = &As[wm * 32 + mi * 16 + l16][hl * 8];
      v8bf lo = *(const v8bf*)ap;         // K = hl*8 .. +7
      v8bf hi = *(const v8bf*)(ap + 16);  // K = 16 + hl*8 .. +7
#pragma unroll
      for (int i = 0; i < 8; ++i) { afr[mi][i] = lo[i]; afr[mi][i + 8] = hi[i]; }
    }
    v16bf bfr[4];
#pragma unroll
    for (int ni = 0; ni < 4; ++ni) {
      const __bf16* bp = &Bs[wn * 64 + ni * 16 + l16][hl * 8];
      v8bf lo = *(const v8bf*)bp;         // K = hl*8 .. +7
      v8bf hi = *(const v8bf*)(bp + 16);  // K = 16 + hl*8 .. +7
#pragma unroll
      for (int i = 0; i < 8; ++i) { bfr[ni][i] = lo[i]; bfr[ni][i + 8] = hi[i]; }
    }

#pragma unroll
    for (int mi = 0; mi < 2; ++mi)
#pragma unroll
      for (int ni = 0; ni < 4; ++ni)
        acc[mi][ni] = __builtin_amdgcn_wmma_f32_16x16x32_bf16(
            false, afr[mi], false, bfr[ni], (short)0, acc[mi][ni], false, false);

    __syncthreads();
  }

  // ---- epilogue: bias + activation, f32 store (C/D layout: VGPR r -> M) ----
#pragma unroll
  for (int mi = 0; mi < 2; ++mi) {
#pragma unroll
    for (int ni = 0; ni < 4; ++ni) {
      int col = n0 + wn * 64 + ni * 16 + l16;
      float bia = bias ? bias[col] : 0.0f;
#pragma unroll
      for (int r = 0; r < 8; ++r) {
        int row = blockIdx.x * BM + wm * 32 + mi * 16 + hl * 8 + r;
        float val = acc[mi][ni][r] + bia;
        if (ACT == 1) val = 0.5f * val * (1.0f + erff(val * 0.70710678118654752f));
        Cg[(size_t)row * ldc + col] = val;
      }
    }
  }
}

// ---------------------------------------------------------------------------
// Embeddings + LayerNorm: x[t] = LN(word_emb[id] + type_emb[0] + pos_emb[s])
// ---------------------------------------------------------------------------
__global__ __launch_bounds__(256)
void embed_ln_kernel(const int* __restrict__ ids,
                     const float* __restrict__ we, const float* __restrict__ pe,
                     const float* __restrict__ te,
                     const float* __restrict__ g, const float* __restrict__ b,
                     float* __restrict__ x, float eps) {
  __shared__ float red[NT];
  const int t   = blockIdx.x;
  const int s   = t % Sc;
  const int id  = ids[t];
  const int tid = threadIdx.x;

  float v[3];
  float part = 0.0f;
#pragma unroll
  for (int j = 0; j < 3; ++j) {
    int d = tid + j * NT;
    v[j] = we[(size_t)id * Dc + d] + te[d] + pe[(size_t)s * Dc + d];
    part += v[j];
  }
  red[tid] = part; __syncthreads();
  for (int st = NT / 2; st > 0; st >>= 1) {
    if (tid < st) red[tid] += red[tid + st];
    __syncthreads();
  }
  float mean = red[0] * (1.0f / Dc); __syncthreads();

  part = 0.0f;
#pragma unroll
  for (int j = 0; j < 3; ++j) { float d0 = v[j] - mean; part += d0 * d0; }
  red[tid] = part; __syncthreads();
  for (int st = NT / 2; st > 0; st >>= 1) {
    if (tid < st) red[tid] += red[tid + st];
    __syncthreads();
  }
  float rstd = rsqrtf(red[0] * (1.0f / Dc) + eps);

#pragma unroll
  for (int j = 0; j < 3; ++j) {
    int d = tid + j * NT;
    x[(size_t)t * Dc + d] = (v[j] - mean) * rstd * g[d] + b[d];
  }
}

// ---------------------------------------------------------------------------
// Residual + LayerNorm (in place on x): x = LN(x + t)
// ---------------------------------------------------------------------------
__global__ __launch_bounds__(256)
void add_ln_kernel(float* __restrict__ x, const float* __restrict__ tbuf,
                   const float* __restrict__ g, const float* __restrict__ b,
                   float eps) {
  __shared__ float red[NT];
  const int t   = blockIdx.x;
  const int tid = threadIdx.x;

  float v[3];
  float part = 0.0f;
#pragma unroll
  for (int j = 0; j < 3; ++j) {
    int d = tid + j * NT;
    v[j] = x[(size_t)t * Dc + d] + tbuf[(size_t)t * Dc + d];
    part += v[j];
  }
  red[tid] = part; __syncthreads();
  for (int st = NT / 2; st > 0; st >>= 1) {
    if (tid < st) red[tid] += red[tid + st];
    __syncthreads();
  }
  float mean = red[0] * (1.0f / Dc); __syncthreads();

  part = 0.0f;
#pragma unroll
  for (int j = 0; j < 3; ++j) { float d0 = v[j] - mean; part += d0 * d0; }
  red[tid] = part; __syncthreads();
  for (int st = NT / 2; st > 0; st >>= 1) {
    if (tid < st) red[tid] += red[tid + st];
    __syncthreads();
  }
  float rstd = rsqrtf(red[0] * (1.0f / Dc) + eps);

#pragma unroll
  for (int j = 0; j < 3; ++j) {
    int d = tid + j * NT;
    x[(size_t)t * Dc + d] = (v[j] - mean) * rstd * g[d] + b[d];
  }
}

// ---------------------------------------------------------------------------
// In-place masked softmax over score rows: row = softmax(row*scale + mask)
// grid = (S, B*H).
// ---------------------------------------------------------------------------
__global__ __launch_bounds__(256)
void softmax_mask_kernel(float* __restrict__ attn, const int* __restrict__ mask,
                         float scale) {
  __shared__ float red[NT];
  const int q   = blockIdx.x;
  const int z   = blockIdx.y;            // b*H + h
  const int bb  = z / Hc;
  const int tid = threadIdx.x;
  float* row = attn + ((size_t)z * Sc + q) * Sc;

  float vals[4];
  float mx = -3.40282347e38f;
#pragma unroll
  for (int j = 0; j < 4; ++j) {
    int c = tid + j * NT;
    float add = (mask[(size_t)bb * Sc + c] != 0) ? 0.0f : -3.40282347e38f;
    vals[j] = row[c] * scale + add;
    mx = fmaxf(mx, vals[j]);
  }
  red[tid] = mx; __syncthreads();
  for (int st = NT / 2; st > 0; st >>= 1) {
    if (tid < st) red[tid] = fmaxf(red[tid], red[tid + st]);
    __syncthreads();
  }
  mx = red[0]; __syncthreads();

  float sum = 0.0f;
#pragma unroll
  for (int j = 0; j < 4; ++j) { vals[j] = __expf(vals[j] - mx); sum += vals[j]; }
  red[tid] = sum; __syncthreads();
  for (int st = NT / 2; st > 0; st >>= 1) {
    if (tid < st) red[tid] += red[tid + st];
    __syncthreads();
  }
  float inv = 1.0f / red[0];

#pragma unroll
  for (int j = 0; j < 4; ++j) row[tid + j * NT] = vals[j] * inv;
}

// ---------------------------------------------------------------------------
// Pooler: pooled[b] = tanh(x[b, 0] @ Wp + bp). Tiny; VALU is fine.
// ---------------------------------------------------------------------------
__global__ __launch_bounds__(256)
void pooler_kernel(const float* __restrict__ x, const float* __restrict__ Wp,
                   const float* __restrict__ bp, float* __restrict__ out) {
  const int bb  = blockIdx.x;
  const int tid = threadIdx.x;
  const float* xr = x + (size_t)bb * Sc * Dc;   // token 0 of batch bb
  for (int n = tid; n < Dc; n += NT) {
    float acc = bp[n];
    for (int k = 0; k < Dc; ++k) acc = fmaf(xr[k], Wp[(size_t)k * Dc + n], acc);
    out[(size_t)bb * Dc + n] = tanhf(acc);
  }
}

// ---------------------------------------------------------------------------
extern "C" void kernel_launch(void* const* d_in, const int* in_sizes, int n_in,
                              void* d_out, int out_size, void* d_ws, size_t ws_size,
                              hipStream_t stream) {
  (void)in_sizes; (void)n_in; (void)out_size; (void)ws_size;

  const int*   input_ids = (const int*)  d_in[0];
  const int*   amask     = (const int*)  d_in[1];
  const float* word_emb  = (const float*)d_in[2];
  const float* pos_emb   = (const float*)d_in[3];
  const float* type_emb  = (const float*)d_in[4];
  const float* emb_g     = (const float*)d_in[5];
  const float* emb_b     = (const float*)d_in[6];
  const float* Wq = (const float*)d_in[7];  const float* bq = (const float*)d_in[8];
  const float* Wk = (const float*)d_in[9];  const float* bk = (const float*)d_in[10];
  const float* Wv = (const float*)d_in[11]; const float* bv = (const float*)d_in[12];
  const float* Wo = (const float*)d_in[13]; const float* bo = (const float*)d_in[14];
  const float* W1 = (const float*)d_in[15]; const float* b1 = (const float*)d_in[16];
  const float* W2 = (const float*)d_in[17]; const float* b2 = (const float*)d_in[18];
  const float* g1 = (const float*)d_in[19]; const float* be1 = (const float*)d_in[20];
  const float* g2 = (const float*)d_in[21]; const float* be2 = (const float*)d_in[22];
  const float* Wp = (const float*)d_in[23]; const float* bp  = (const float*)d_in[24];

  float* outp   = (float*)d_out;
  float* pooled = outp;                       // [B, D]
  float* attn   = outp + (size_t)Bc * Dc;     // [L, B, H, S, S]

  const size_t MD = (size_t)Bc * Sc * Dc;     // 8192 * 768
  const size_t DD = (size_t)Dc * Dc;
  const size_t DF = (size_t)Dc * FFc;
  float* ws  = (float*)d_ws;
  float* x   = ws;                            // hidden states  [B*S, D]
  float* q   = ws + 1 * MD;
  float* k   = ws + 2 * MD;
  float* v   = ws + 3 * MD;
  float* ctx = ws + 4 * MD;
  float* t   = ws + 5 * MD;                   // proj / ffn2 output
  float* ff  = ws + 6 * MD;                   // [B*S, FF]
  __bf16* wtq = (__bf16*)(ws + 6 * MD + (size_t)Bc * Sc * FFc);
  __bf16* wtk = wtq + DD;
  __bf16* wtv = wtk + DD;
  __bf16* wto = wtv + DD;
  __bf16* wt1 = wto + DD;                     // [FF][D] -> bf16 [N=FF][K=D]
  __bf16* wt2 = wt1 + DF;                     // [D][FF] -> bf16 [N=D][K=FF]

  const dim3 blk(NT);
  const int M  = Bc * Sc;                     // 8192
  const long SD  = (long)Sc * Dc;
  const long HSS = (long)Hc * Sc * Sc;
  const long SS  = (long)Sc * Sc;

  embed_ln_kernel<<<dim3(M), blk, 0, stream>>>(
      input_ids, word_emb, pos_emb, type_emb, emb_g, emb_b, x, 1e-12f);

  for (int l = 0; l < Lc; ++l) {
    const float* bq_l = bq + (size_t)l * Dc;
    const float* bk_l = bk + (size_t)l * Dc;
    const float* bv_l = bv + (size_t)l * Dc;
    const float* bo_l = bo + (size_t)l * Dc;
    const float* b1_l = b1 + (size_t)l * FFc;
    const float* b2_l = b2 + (size_t)l * Dc;
    float* attnL = attn + (size_t)l * Bc * HSS;

    // ---- one-time per-layer weight prepack: f32 [K][N] -> bf16 [N][K] ----
    prepack_wT_kernel<<<dim3(DD / NT), blk, 0, stream>>>(Wq + l * DD, wtq, Dc, Dc);
    prepack_wT_kernel<<<dim3(DD / NT), blk, 0, stream>>>(Wk + l * DD, wtk, Dc, Dc);
    prepack_wT_kernel<<<dim3(DD / NT), blk, 0, stream>>>(Wv + l * DD, wtv, Dc, Dc);
    prepack_wT_kernel<<<dim3(DD / NT), blk, 0, stream>>>(Wo + l * DD, wto, Dc, Dc);
    prepack_wT_kernel<<<dim3(DF / NT), blk, 0, stream>>>(W1 + l * DF, wt1, Dc, FFc);
    prepack_wT_kernel<<<dim3(DF / NT), blk, 0, stream>>>(W2 + l * DF, wt2, FFc, Dc);

    // Q, K, V projections: [8192,768] @ [768,768] + bias (bf16 weights, async)
    gemm_bf16_wmma<4, 2, 2, 0><<<dim3(M / 128, Dc / 128, 1), blk, 0, stream>>>(
        x, Dc, 0, 0, wtq, Dc, 0, 0, q, Dc, 0, 0, bq_l, Dc, 1);
    gemm_bf16_wmma<4, 2, 2, 0><<<dim3(M / 128, Dc / 128, 1), blk, 0, stream>>>(
        x, Dc, 0, 0, wtk, Dc, 0, 0, k, Dc, 0, 0, bk_l, Dc, 1);
    gemm_bf16_wmma<4, 2, 2, 0><<<dim3(M / 128, Dc / 128, 1), blk, 0, stream>>>(
        x, Dc, 0, 0, wtv, Dc, 0, 0, v, Dc, 0, 0, bv_l, Dc, 1);

    // scores[b,h] = Q[b,h] @ K[b,h]^T  -> written straight into d_out
    gemm_bf16_wmma<4, 2, 1, 0><<<dim3(Sc / 128, Sc / 128, Bc * Hc), blk, 0, stream>>>(
        q, Dc, SD, DKc, k, Dc, SD, DKc, attnL, Sc, HSS, SS, nullptr, DKc, Hc);

    // softmax(scale + mask) in place
    softmax_mask_kernel<<<dim3(Sc, Bc * Hc), blk, 0, stream>>>(attnL, amask, 0.125f);

    // ctx[b,h] = attn[b,h] @ V[b,h]  (M=1024, N=64, K=1024) -> [B*S, D] slices
    gemm_bf16_wmma<8, 1, 0, 0><<<dim3(Sc / 256, 1, Bc * Hc), blk, 0, stream>>>(
        attnL, Sc, HSS, SS, v, Dc, SD, DKc, ctx, Dc, SD, DKc, nullptr, Sc, Hc);

    // output projection + residual LN
    gemm_bf16_wmma<4, 2, 2, 0><<<dim3(M / 128, Dc / 128, 1), blk, 0, stream>>>(
        ctx, Dc, 0, 0, wto, Dc, 0, 0, t, Dc, 0, 0, bo_l, Dc, 1);
    add_ln_kernel<<<dim3(M), blk, 0, stream>>>(x, t, g1 + (size_t)l * Dc,
                                               be1 + (size_t)l * Dc, 1e-5f);

    // FFN: GELU(x @ W1 + b1) @ W2 + b2, residual LN
    gemm_bf16_wmma<4, 2, 2, 1><<<dim3(M / 128, FFc / 128, 1), blk, 0, stream>>>(
        x, Dc, 0, 0, wt1, Dc, 0, 0, ff, FFc, 0, 0, b1_l, Dc, 1);
    gemm_bf16_wmma<4, 2, 2, 0><<<dim3(M / 128, Dc / 128, 1), blk, 0, stream>>>(
        ff, FFc, 0, 0, wt2, FFc, 0, 0, t, Dc, 0, 0, b2_l, FFc, 1);
    add_ln_kernel<<<dim3(M), blk, 0, stream>>>(x, t, g2 + (size_t)l * Dc,
                                               be2 + (size_t)l * Dc, 1e-5f);
  }

  pooler_kernel<<<dim3(Bc), blk, 0, stream>>>(x, Wp, bp, pooled);
}
